// Attention_67637144977874
// MI455X (gfx1250) — matrix operand order repaired and checked by reading, not compiled, not run on hardware
//
#include <hip/hip_runtime.h>
#include <hip/hip_bf16.h>
#include <cstdint>

typedef _Float16 v16h __attribute__((ext_vector_type(16)));
typedef _Float16 v8h  __attribute__((ext_vector_type(8)));
typedef _Float16 v4h  __attribute__((ext_vector_type(4)));
typedef float    v8f  __attribute__((ext_vector_type(8)));

#define DIM 512
#define NHEADS 8
#define NTOK 256
#define BATCH 256
#define HD 64
#define QKV_COLS 1536
#define MROWS (BATCH * NTOK)          // 65536
#define SCALE 0.125f                  // 64^-0.5

static __device__ __forceinline__ v16h mk16(v8h lo, v8h hi) {
    return __builtin_shufflevector(lo, hi, 0,1,2,3,4,5,6,7,8,9,10,11,12,13,14,15);
}

// LDS byte offset of a generic pointer that points into LDS
static __device__ __forceinline__ uint32_t lds_off(const _Float16* p) {
    return (uint32_t)(uintptr_t)(__attribute__((address_space(3))) const _Float16*)p;
}

// CDNA5 async global->LDS copy, 16B per lane, tracked by ASYNCcnt (no VGPR transit)
static __device__ __forceinline__ void async_cp16(uint32_t ldsoff, uint64_t gaddr) {
    asm volatile("global_load_async_to_lds_b128 %0, %1, off"
                 :: "v"(ldsoff), "v"(gaddr) : "memory");
}
static __device__ __forceinline__ void wait_async0() {
    asm volatile("s_wait_asynccnt 0x0" ::: "memory");
}

// ---------------------------------------------------------------- prep kernels
__global__ void prep_x_kernel(const float* __restrict__ x, const float* __restrict__ noise,
                              const float* __restrict__ ns, _Float16* __restrict__ xh, int n4) {
    int i = blockIdx.x * blockDim.x + threadIdx.x;
    if (i >= n4) return;
    float s = ns[0];
    float4 xv = ((const float4*)x)[i];
    float nz = noise[(i * 4) >> 9] * s;
    v4h o;
    o[0] = (_Float16)(xv.x + nz); o[1] = (_Float16)(xv.y + nz);
    o[2] = (_Float16)(xv.z + nz); o[3] = (_Float16)(xv.w + nz);
    ((v4h*)xh)[i] = o;
}

__global__ void prep_wqkv_kernel(const float* __restrict__ w, _Float16* __restrict__ wt, int n) {
    int i = blockIdx.x * blockDim.x + threadIdx.x;
    if (i >= n) return;
    int c = i >> 9, k = i & 511;                 // wt[c][k] = w[k][c]
    wt[i] = (_Float16)w[(size_t)k * QKV_COLS + c];
}

__global__ void prep_wproj_kernel(const float* __restrict__ w, _Float16* __restrict__ wt, int n) {
    int i = blockIdx.x * blockDim.x + threadIdx.x;
    if (i >= n) return;
    int c = i >> 9, k = i & 511;
    wt[i] = (_Float16)w[(size_t)k * DIM + c];
}

// bias stored TRANSPOSED: biasT[h][j][i] so the attention C-operand column is contiguous
__global__ void prep_bias_kernel(const float* __restrict__ table, const int* __restrict__ ridx,
                                 float* __restrict__ biasT, int n) {
    int idx = blockIdx.x * blockDim.x + threadIdx.x;
    if (idx >= n) return;
    int h = idx >> 16, j = (idx >> 8) & 255, i = idx & 255;
    biasT[idx] = table[ridx[i * NTOK + j] * NHEADS + h];
}

// ------------------------------------------------------------ shared GEMM core
// C[128x128] tile at (m0,n0): A [M,512] f16 row-major, B [Ncols,512] f16 (pre-transposed).
// K = 512. 4 waves, each 64x64 = 4x4 WMMA tiles. Double-buffered LDS, async staging.
#define LDS_STRIDE 40                              // 32 + 8 pad halves
#define GEMM_BUF   (2 * 128 * LDS_STRIDE)          // one A+B buffer, in halves

__device__ __forceinline__ void gemm_stage(const _Float16* A, const _Float16* B,
                                           int m0, int n0, int k0, _Float16* lds, int buf) {
    const int t = threadIdx.x;
    _Float16* base = lds + buf * GEMM_BUF;
    uint32_t la = lds_off(base + t * LDS_STRIDE);
    uint64_t ga = (uint64_t)(uintptr_t)(A + (size_t)(m0 + t) * DIM + k0);
    uint32_t lb = lds_off(base + 128 * LDS_STRIDE + t * LDS_STRIDE);
    uint64_t gb = (uint64_t)(uintptr_t)(B + (size_t)(n0 + t) * DIM + k0);
    #pragma unroll
    for (int i = 0; i < 4; ++i) async_cp16(la + i * 16, ga + i * 16);
    #pragma unroll
    for (int i = 0; i < 4; ++i) async_cp16(lb + i * 16, gb + i * 16);
}

__device__ __forceinline__ void gemm_tiles(const _Float16* __restrict__ A,
                                           const _Float16* __restrict__ B,
                                           int m0, int n0, _Float16* lds, v8f acc[4][4]) {
    const int t = threadIdx.x, l = t & 31, w = t >> 5;
    const int wm = (w >> 1) * 64, wn = (w & 1) * 64;
    v8f z = {};
    #pragma unroll
    for (int mi = 0; mi < 4; ++mi)
        #pragma unroll
        for (int ni = 0; ni < 4; ++ni) acc[mi][ni] = z;

    gemm_stage(A, B, m0, n0, 0, lds, 0);
    int cur = 0;
    for (int k0 = 0; k0 < DIM; k0 += 32) {
        wait_async0();          // own async writes landed in LDS
        __syncthreads();        // everyone's writes visible; prev buffer free for refill
        if (k0 + 32 < DIM) gemm_stage(A, B, m0, n0, k0 + 32, lds, cur ^ 1);

        const _Float16* bb = lds + cur * GEMM_BUF;
        v16h af[4], bf[4];
        const int kh = (l >> 4) * 8;
        #pragma unroll
        for (int mi = 0; mi < 4; ++mi) {
            const _Float16* p = bb + (wm + mi * 16 + (l & 15)) * LDS_STRIDE;
            af[mi] = mk16(*(const v8h*)(p + kh), *(const v8h*)(p + 16 + kh));
        }
        const int ks = (l >> 4) * 8;
        #pragma unroll
        for (int ni = 0; ni < 4; ++ni) {
            const _Float16* p = bb + 128 * LDS_STRIDE + (wn + ni * 16 + (l & 15)) * LDS_STRIDE + ks;
            bf[ni] = mk16(*(const v8h*)p, *(const v8h*)(p + 16));
        }
        #pragma unroll
        for (int mi = 0; mi < 4; ++mi)
            #pragma unroll
            for (int ni = 0; ni < 4; ++ni)
                acc[mi][ni] = __builtin_amdgcn_wmma_f32_16x16x32_f16(
                    false, af[mi], false, bf[ni], (short)0, acc[mi][ni], false, false);
        cur ^= 1;
    }
}

// ----------------------------------------------------- GEMM 1: x @ Wqkv -> Q,K,Vt
__global__ void __launch_bounds__(128)
gemm_qkv_kernel(const _Float16* __restrict__ A, const _Float16* __restrict__ B,
                _Float16* __restrict__ Q, _Float16* __restrict__ K, _Float16* __restrict__ Vt) {
    __shared__ __align__(16) _Float16 lds[2 * GEMM_BUF];
    const int m0 = blockIdx.y * 128, n0 = blockIdx.x * 128;
    v8f acc[4][4];
    gemm_tiles(A, B, m0, n0, lds, acc);

    const int t = threadIdx.x, l = t & 31, w = t >> 5;
    const int wm = (w >> 1) * 64, wn = (w & 1) * 64;
    #pragma unroll
    for (int mi = 0; mi < 4; ++mi) {
        #pragma unroll
        for (int ni = 0; ni < 4; ++ni) {
            const int colb = n0 + wn + ni * 16;
            const int which = colb >> 9;
            const int head  = (colb >> 6) & 7;
            const int hd    = (colb & 63) + (l & 15);
            const int rowb  = m0 + wm + mi * 16 + 8 * (l >> 4);
            const int b = rowb >> 8, tok0 = rowb & 255;
            if (which == 2) {
                // 8 consecutive tokens per lane -> one 16B store into Vt [b,h,hd,n]
                union { uint4 u4; _Float16 h[8]; } pk;
                #pragma unroll
                for (int r = 0; r < 8; ++r) pk.h[r] = (_Float16)acc[mi][ni][r];
                *(uint4*)(Vt + (((size_t)(b * NHEADS + head) * HD + hd) * NTOK) + tok0) = pk.u4;
            } else {
                _Float16* dst = (which == 0) ? Q : K;
                const float sc = (which == 0) ? SCALE : 1.f;
                #pragma unroll
                for (int r = 0; r < 8; ++r)
                    dst[(((size_t)(b * NHEADS + head) * NTOK + tok0 + r) * HD) + hd] =
                        (_Float16)(acc[mi][ni][r] * sc);
            }
        }
    }
}

// ------------------------------------------------------------------- attention
// block = 128 threads (4 waves), grid = (4 qblocks, 8 heads, 256 windows).
// LDS: Ks[256][72] + Vts[64][264] + Ps[4][16][264] (f16) = 104448 B (dynamic).
#define KS_STRIDE 72
#define VT_STRIDE 264
__global__ void __launch_bounds__(128)
attn_kernel(const _Float16* __restrict__ Q, const _Float16* __restrict__ K,
            const _Float16* __restrict__ Vt, const float* __restrict__ biasT,
            _Float16* __restrict__ O) {
    extern __shared__ __align__(16) _Float16 smem[];
    _Float16* Ks  = smem;                          // 256*72
    _Float16* Vts = smem + 256 * KS_STRIDE;        // 64*264
    const int t = threadIdx.x, l = t & 31, w = t >> 5;
    _Float16* Ps = smem + 256 * KS_STRIDE + 64 * VT_STRIDE + w * 16 * VT_STRIDE;

    const int b = blockIdx.z, h = blockIdx.y, qb = blockIdx.x;
    const size_t bh = (size_t)(b * NHEADS + h);

    // async-stage K [tok][hd] and Vt [hd][tok] into LDS
    {
        const _Float16* Kg = K + bh * NTOK * HD;
        #pragma unroll
        for (int rep = 0; rep < 2; ++rep) {
            const int rr = t + rep * 128;
            uint32_t lk = lds_off(Ks + rr * KS_STRIDE);
            uint64_t gk = (uint64_t)(uintptr_t)(Kg + (size_t)rr * HD);
            #pragma unroll
            for (int i = 0; i < 8; ++i) async_cp16(lk + i * 16, gk + i * 16);
        }
        const _Float16* Vg = Vt + bh * HD * NTOK;
        const int vrow = t >> 1, hoff = (t & 1) * 128;
        uint32_t lv = lds_off(Vts + vrow * VT_STRIDE + hoff);
        uint64_t gv = (uint64_t)(uintptr_t)(Vg + (size_t)vrow * NTOK + hoff);
        #pragma unroll
        for (int i = 0; i < 16; ++i) async_cp16(lv + i * 16, gv + i * 16);
        wait_async0();
    }
    __syncthreads();

    // Q A-fragments for this wave's 16 query rows (already scaled by 1/sqrt(hd))
    const int q0 = qb * 64 + w * 16;
    const _Float16* qrow = Q + (bh * NTOK + q0 + (l & 15)) * HD;
    const int kh = (l >> 4) * 8;
    v16h qf0 = mk16(*(const v8h*)(qrow + kh),      *(const v8h*)(qrow + 16 + kh));
    v16h qf1 = mk16(*(const v8h*)(qrow + 32 + kh), *(const v8h*)(qrow + 48 + kh));

    // S = Q K^T + bias : 16 tiles of 16x16; transposed bias rides in as C (one v8f load)
    v8f s[16];
    const float* bg = biasT + (size_t)h * NTOK * NTOK;
    const int rb = q0 + 8 * (l >> 4);
    const int ks = (l >> 4) * 8;
    #pragma unroll
    for (int jt = 0; jt < 16; ++jt) {
        const int colj = jt * 16 + (l & 15);
        v8f c = *(const v8f*)(bg + (size_t)colj * NTOK + rb);
        const _Float16* kp = Ks + (jt * 16 + (l & 15)) * KS_STRIDE;
        v16h kf0 = mk16(*(const v8h*)(kp + ks),      *(const v8h*)(kp + ks + 16));
        v16h kf1 = mk16(*(const v8h*)(kp + 32 + ks), *(const v8h*)(kp + 32 + ks + 16));
        c = __builtin_amdgcn_wmma_f32_16x16x32_f16(false, qf0, false, kf0, (short)0, c, false, false);
        c = __builtin_amdgcn_wmma_f32_16x16x32_f16(false, qf1, false, kf1, (short)0, c, false, false);
        s[jt] = c;
    }

    // row softmax: each row lives in one VGPR slot across a 16-lane half
    float rsum[8];
    #pragma unroll
    for (int r = 0; r < 8; ++r) {
        float m = -1e30f;
        #pragma unroll
        for (int jt = 0; jt < 16; ++jt) m = fmaxf(m, s[jt][r]);
        #pragma unroll
        for (int off = 1; off < 16; off <<= 1) m = fmaxf(m, __shfl_xor(m, off));
        float sum = 0.f;
        #pragma unroll
        for (int jt = 0; jt < 16; ++jt) {
            float e = __expf(s[jt][r] - m);
            s[jt][r] = e;
            sum += e;
        }
        #pragma unroll
        for (int off = 1; off < 16; off <<= 1) sum += __shfl_xor(sum, off);
        rsum[r] = sum;
    }

    // C-layout -> A-layout reshape through LDS
    #pragma unroll
    for (int jt = 0; jt < 16; ++jt)
        #pragma unroll
        for (int r = 0; r < 8; ++r)
            Ps[(r + 8 * (l >> 4)) * VT_STRIDE + jt * 16 + (l & 15)] = (_Float16)s[jt][r];
    __syncthreads();

    // O = P V : 4 hd-tiles, 8 token chunks of 32
    v8f o[4];
    v8f z = {};
    #pragma unroll
    for (int ni = 0; ni < 4; ++ni) o[ni] = z;
    for (int kc = 0; kc < 8; ++kc) {
        const _Float16* pp = Ps + (l & 15) * VT_STRIDE + kc * 32;
        v16h pf = mk16(*(const v8h*)(pp + kh), *(const v8h*)(pp + 16 + kh));
        #pragma unroll
        for (int ni = 0; ni < 4; ++ni) {
            const _Float16* vp = Vts + (ni * 16 + (l & 15)) * VT_STRIDE + kc * 32 + ks;
            v16h vf = mk16(*(const v8h*)vp, *(const v8h*)(vp + 16));
            o[ni] = __builtin_amdgcn_wmma_f32_16x16x32_f16(false, pf, false, vf, (short)0, o[ni], false, false);
        }
    }

    // normalize and store [bs, n, dim] f16
    #pragma unroll
    for (int ni = 0; ni < 4; ++ni) {
        const int col = h * HD + ni * 16 + (l & 15);
        #pragma unroll
        for (int r = 0; r < 8; ++r) {
            const int row = b * NTOK + q0 + r + 8 * (l >> 4);
            O[(size_t)row * DIM + col] = (_Float16)(o[ni][r] / rsum[r]);
        }
    }
}

// ----------------------------------------------------- GEMM 2: O @ Wproj + bias
__global__ void __launch_bounds__(128)
gemm_proj_kernel(const _Float16* __restrict__ A, const _Float16* __restrict__ B,
                 const float* __restrict__ bias, float* __restrict__ out) {
    __shared__ __align__(16) _Float16 lds[2 * GEMM_BUF];
    const int m0 = blockIdx.y * 128, n0 = blockIdx.x * 128;
    v8f acc[4][4];
    gemm_tiles(A, B, m0, n0, lds, acc);

    const int t = threadIdx.x, l = t & 31, w = t >> 5;
    const int wm = (w >> 1) * 64, wn = (w & 1) * 64;
    #pragma unroll
    for (int mi = 0; mi < 4; ++mi) {
        #pragma unroll
        for (int ni = 0; ni < 4; ++ni) {
            const int col = n0 + wn + ni * 16 + (l & 15);
            const float bp = bias[col];
            const int rowb = m0 + wm + mi * 16 + 8 * (l >> 4);
            #pragma unroll
            for (int r = 0; r < 8; ++r)
                out[(size_t)(rowb + r) * DIM + col] = acc[mi][ni][r] + bp;
        }
    }
}

// -------------------------------------------------------------------- launcher
extern "C" void kernel_launch(void* const* d_in, const int* in_sizes, int n_in,
                              void* d_out, int out_size, void* d_ws, size_t ws_size,
                              hipStream_t stream) {
    const float* x      = (const float*)d_in[0];
    const float* noise  = (const float*)d_in[1];
    const float* nstr   = (const float*)d_in[2];
    const float* w_qkv  = (const float*)d_in[3];
    const float* w_proj = (const float*)d_in[4];
    const float* b_proj = (const float*)d_in[5];
    const float* btab   = (const float*)d_in[6];
    const int*   ridx   = (const int*)d_in[7];
    float* out = (float*)d_out;

    char* ws = (char*)d_ws;
    const size_t SZ_XH    = (size_t)MROWS * DIM * 2;             // 64 MB (reused as O)
    const size_t SZ_WQKV  = (size_t)QKV_COLS * DIM * 2;          // 1.5 MB
    const size_t SZ_WPROJ = (size_t)DIM * DIM * 2;               // 0.5 MB
    const size_t SZ_QKV   = (size_t)BATCH * NHEADS * NTOK * HD * 2; // 64 MB each
    _Float16* xh     = (_Float16*)(ws);
    _Float16* wqkvT  = (_Float16*)(ws + SZ_XH);
    _Float16* wprojT = (_Float16*)(ws + SZ_XH + SZ_WQKV);
    _Float16* Qh     = (_Float16*)(ws + SZ_XH + SZ_WQKV + SZ_WPROJ);
    _Float16* Kh     = (_Float16*)((char*)Qh + SZ_QKV);
    _Float16* Vt     = (_Float16*)((char*)Kh + SZ_QKV);
    float*    biasT  = (float*)((char*)Vt + SZ_QKV);             // 2 MB [H,N,N] transposed

    // 1) prep
    {
        int n4 = (MROWS * DIM) / 4;
        prep_x_kernel<<<(n4 + 255) / 256, 256, 0, stream>>>(x, noise, nstr, xh, n4);
    }
    {
        int n = QKV_COLS * DIM;
        prep_wqkv_kernel<<<(n + 255) / 256, 256, 0, stream>>>(w_qkv, wqkvT, n);
    }
    {
        int n = DIM * DIM;
        prep_wproj_kernel<<<(n + 255) / 256, 256, 0, stream>>>(w_proj, wprojT, n);
    }
    {
        int n = NHEADS * NTOK * NTOK;
        prep_bias_kernel<<<(n + 255) / 256, 256, 0, stream>>>(btab, ridx, biasT, n);
    }

    // 2) QKV GEMM: [65536,512] x [512,1536]
    gemm_qkv_kernel<<<dim3(QKV_COLS / 128, MROWS / 128), 128, 0, stream>>>(xh, wqkvT, Qh, Kh, Vt);

    // 3) attention (reuses xh as O buffer; xh is dead after the QKV GEMM)
    {
        size_t smem = (size_t)(256 * KS_STRIDE + 64 * VT_STRIDE + 4 * 16 * VT_STRIDE) * 2;
        attn_kernel<<<dim3(NTOK / 64, NHEADS, BATCH), 128, smem, stream>>>(Qh, Kh, Vt, biasT, xh);
    }

    // 4) projection GEMM: [65536,512] x [512,512] + bias
    gemm_proj_kernel<<<dim3(DIM / 128, MROWS / 128), 128, 0, stream>>>(xh, wprojT, b_proj, out);
}
